// PCAModel_21560735825971
// MI455X (gfx1250) — compile-verified, matrix operand order checked by reading, not run on hardware
//
#include <hip/hip_runtime.h>
#include <stdint.h>
#include <stddef.h>

// ---------------------------------------------------------------------------
// PCA-GM pipeline for MI455X (gfx1250, wave32, WMMA bf16 16x16x32)
// ---------------------------------------------------------------------------

typedef __attribute__((ext_vector_type(16))) __bf16 bf16x16;
typedef __attribute__((ext_vector_type(8)))  float  f32x8;

#define NN        4096      // N1 == N2
#define DIN       64
#define DH        128
#define NE        65536
#define SKITERS   5
#define SKEPS     1e-6f
#define CLAMPMAX  85.0f

static __device__ __forceinline__ uint16_t f2bf(float f) {
  union { float f; uint32_t u; } v; v.f = f;
  uint32_t u = v.u;
  uint32_t r = u + 0x7fffu + ((u >> 16) & 1u);   // round-to-nearest-even
  return (uint16_t)(r >> 16);
}
static __device__ __forceinline__ float bf2f(uint16_t h) {
  union { float f; uint32_t u; } v; v.u = ((uint32_t)h) << 16;
  return v.f;
}

// ---------------------------------------------------------------------------
// WMMA GEMM:  C[M x N] = epilogue( A[MxK](bf16) @ W[NxK](bf16)^T )
// One wave -> 16x32 output tile (two 16x16 WMMA tiles sharing the A fragment).
// Fragment layout (contiguous K, not the ISA table order): lane holds 16 consecutive K of row (lane&15),
// K-half selected by lane>>4 -- identical pattern for A and for W rows (=B cols).
// Epilogue modes:
//   0: f32 = x
//   1: f32 = relu(x + bias[n])
//   2: bf16 = exp(min(x*scale, 85))            (affinity)
//   3: bf16 = x * rowscale[row]  (+ column offset into concat buffer)
//   4: bf16 = x
//   6: bf16 = relu(x + bias[n])
// ---------------------------------------------------------------------------
__global__ void gemm_bf16_wmma(const uint16_t* __restrict__ A,
                               const uint16_t* __restrict__ Wt,
                               int M, int N, int K,
                               int mode, float scale,
                               const float* __restrict__ bias,
                               const float* __restrict__ rowscale,
                               float* __restrict__ outF, int ldF,
                               uint16_t* __restrict__ outB, int ldB, int colOffB)
{
  const int lane = threadIdx.x & 31;
  const int wave = blockIdx.x * (blockDim.x >> 5) + (threadIdx.x >> 5);
  const int nt2  = N >> 5;                 // N must be a multiple of 32
  const int tm   = wave / nt2;
  const int tn2  = wave - tm * nt2;
  if (tm >= (M >> 4)) return;              // wave-uniform: EXEC stays all-ones

  const int rowA  = (tm << 4) + (lane & 15);
  const int rowW0 = (tn2 << 5) + (lane & 15);
  const int koff  = (lane >> 4) << 4;      // 0 or 16

  const uint16_t* ap = A  + (size_t)rowA  * K + koff;
  const uint16_t* w0 = Wt + (size_t)rowW0 * K + koff;
  const uint16_t* w1 = w0 + ((size_t)K << 4);   // +16 rows of W

  f32x8 acc0 = {0.f,0.f,0.f,0.f,0.f,0.f,0.f,0.f};
  f32x8 acc1 = {0.f,0.f,0.f,0.f,0.f,0.f,0.f,0.f};

  #pragma unroll 4
  for (int k = 0; k < K; k += 32) {
    bf16x16 av = *reinterpret_cast<const bf16x16*>(ap + k);
    bf16x16 b0 = *reinterpret_cast<const bf16x16*>(w0 + k);
    bf16x16 b1 = *reinterpret_cast<const bf16x16*>(w1 + k);
    acc0 = __builtin_amdgcn_wmma_f32_16x16x32_bf16(false, av, false, b0,
                                                   (short)0, acc0, false, false);
    acc1 = __builtin_amdgcn_wmma_f32_16x16x32_bf16(false, av, false, b1,
                                                   (short)0, acc1, false, false);
  }

  // C/D layout: VGPR v, lane L -> row = 16*tm + v + 8*(L>>4), col = L&15
  const int rBase = (tm << 4) + ((lane >> 4) << 3);
  const int c0 = (tn2 << 5) + (lane & 15);
  const int c1 = c0 + 16;
  #pragma unroll
  for (int v = 0; v < 8; ++v) {
    const int r = rBase + v;
    float x0 = acc0[v];
    float x1 = acc1[v];
    if (mode == 0) {
      outF[(size_t)r * ldF + c0] = x0;
      outF[(size_t)r * ldF + c1] = x1;
    } else if (mode == 1) {
      outF[(size_t)r * ldF + c0] = fmaxf(x0 + bias[c0], 0.f);
      outF[(size_t)r * ldF + c1] = fmaxf(x1 + bias[c1], 0.f);
    } else if (mode == 2) {
      outB[(size_t)r * ldB + c0] = f2bf(__expf(fminf(x0 * scale, CLAMPMAX)));
      outB[(size_t)r * ldB + c1] = f2bf(__expf(fminf(x1 * scale, CLAMPMAX)));
    } else if (mode == 3) {
      const float s = rowscale[r];
      outB[(size_t)r * ldB + colOffB + c0] = f2bf(x0 * s);
      outB[(size_t)r * ldB + colOffB + c1] = f2bf(x1 * s);
    } else if (mode == 4) {
      outB[(size_t)r * ldB + c0] = f2bf(x0);
      outB[(size_t)r * ldB + c1] = f2bf(x1);
    } else { // 6
      outB[(size_t)r * ldB + c0] = f2bf(fmaxf(x0 + bias[c0], 0.f));
      outB[(size_t)r * ldB + c1] = f2bf(fmaxf(x1 + bias[c1], 0.f));
    }
  }
}

// ---------------------------------------------------------------------------
// Helpers
// ---------------------------------------------------------------------------
__global__ void f32_to_bf16_k(const float* __restrict__ in,
                              uint16_t* __restrict__ out, int n) {
  int i = blockIdx.x * blockDim.x + threadIdx.x;
  if (i < n) out[i] = f2bf(in[i]);
}

// copy [rows x 128] f32 -> bf16 at column offset inside a wider row
__global__ void f32_to_bf16_strided_k(const float* __restrict__ in, int ldi,
                                      uint16_t* __restrict__ out, int ldo, int colOff) {
  int r = blockIdx.x, c = threadIdx.x;
  out[(size_t)r * ldo + colOff + c] = f2bf(in[(size_t)r * ldi + c]);
}

__global__ void fill_f32_k(float* __restrict__ p, float v, int n) {
  int i = blockIdx.x * blockDim.x + threadIdx.x;
  if (i < n) p[i] = v;
}

// per-edge scatter: agg[dst] += H[src], cnt[dst] += 1   (D = 128 channels)
__global__ void scatter_mean_k(const float* __restrict__ H,
                               const int* __restrict__ edge,
                               float* __restrict__ agg, float* __restrict__ cnt,
                               int E, int D) {
  int e = blockIdx.x, c = threadIdx.x;
  int s = edge[e];
  int d = edge[E + e];
  atomicAdd(&agg[(size_t)d * D + c], H[(size_t)s * D + c]);
  if (c == 0) atomicAdd(&cnt[d], 1.0f);
}

__global__ void gconv_combine_k(const float* __restrict__ Hn,
                                const float* __restrict__ agg,
                                const float* __restrict__ cnt,
                                float* __restrict__ out, int D) {
  int r = blockIdx.x, c = threadIdx.x;
  out[(size_t)r * D + c] = Hn[(size_t)r * D + c]
                         + agg[(size_t)r * D + c] / fmaxf(cnt[r], 1.0f);
}

// row-wise L2 normalize (D = 128) -> bf16
__global__ void rownorm_bf16_k(const float* __restrict__ f,
                               uint16_t* __restrict__ out, int D) {
  int r = blockIdx.x, t = threadIdx.x;
  float v = f[(size_t)r * D + t];
  __shared__ float red[128];
  red[t] = v * v; __syncthreads();
  for (int s = 64; s > 0; s >>= 1) { if (t < s) red[t] += red[t + s]; __syncthreads(); }
  float inv = 1.0f / fmaxf(sqrtf(red[0]), 1e-12f);
  out[(size_t)r * D + t] = f2bf(v * inv);
}

__global__ void scale_rows_bf16_k(const float* __restrict__ f,
                                  const float* __restrict__ s,
                                  uint16_t* __restrict__ out, int D) {
  int r = blockIdx.x, c = threadIdx.x;
  out[(size_t)r * D + c] = f2bf(f[(size_t)r * D + c] * s[r]);
}

// bf16 tiled transpose, block (32,8), grid (cols/32, rows/32)
__global__ void transpose_bf16_k(const uint16_t* __restrict__ in,
                                 uint16_t* __restrict__ out, int rows, int cols) {
  __shared__ uint16_t tile[32][33];
  int bx = blockIdx.x * 32, by = blockIdx.y * 32;
  int tx = threadIdx.x, ty = threadIdx.y;
  for (int i = ty; i < 32; i += 8)
    tile[i][tx] = in[(size_t)(by + i) * cols + (bx + tx)];
  __syncthreads();
  for (int i = ty; i < 32; i += 8)
    out[(size_t)(bx + i) * rows + (by + tx)] = tile[tx][i];
}

// ---- Sinkhorn with scaling vectors: S = diag(r) * M * diag(c) ----
// row step:  r_i <- r_i / max(r_i * sum_j M_ij c_j, eps)
__global__ void sk_rowpass_k(const uint16_t* __restrict__ M,
                             const float* __restrict__ c,
                             float* __restrict__ r, int N) {
  int row = blockIdx.x, t = threadIdx.x;   // 128 threads
  const uint16_t* mr = M + (size_t)row * N;
  float acc = 0.f;
  for (int j = t; j < N; j += 128) acc += bf2f(mr[j]) * c[j];
  __shared__ float red[128];
  red[t] = acc; __syncthreads();
  for (int s = 64; s > 0; s >>= 1) { if (t < s) red[t] += red[t + s]; __syncthreads(); }
  if (t == 0) { float rv = r[row]; r[row] = rv / fmaxf(rv * red[0], SKEPS); }
}

// col partial sums: tcol_j += sum_{i in chunk} r_i M_ij   (coalesced reads)
__global__ void sk_colsum_k(const uint16_t* __restrict__ M,
                            const float* __restrict__ r,
                            float* __restrict__ tcol, int N) {
  int col = blockIdx.x * blockDim.x + threadIdx.x;
  int i0 = blockIdx.y * 256;
  float acc = 0.f;
  for (int i = i0; i < i0 + 256; ++i) acc += r[i] * bf2f(M[(size_t)i * N + col]);
  atomicAdd(&tcol[col], acc);
}

__global__ void sk_colfin_k(float* __restrict__ c,
                            const float* __restrict__ tcol, int N) {
  int j = blockIdx.x * blockDim.x + threadIdx.x;
  if (j < N) { float cv = c[j]; c[j] = cv / fmaxf(cv * tcol[j], SKEPS); }
}

__global__ void sk_materialize_k(const uint16_t* __restrict__ M,
                                 const float* __restrict__ r,
                                 const float* __restrict__ c,
                                 float* __restrict__ out, int N) {
  int col = blockIdx.x * blockDim.x + threadIdx.x;
  int row = blockIdx.y;
  out[(size_t)row * N + col] = r[row] * bf2f(M[(size_t)row * N + col]) * c[col];
}

// ---------------------------------------------------------------------------
// Host orchestration
// ---------------------------------------------------------------------------
static inline void launch_gemm(hipStream_t st,
                               const uint16_t* A, const uint16_t* W,
                               int M, int N, int K, int mode, float scale,
                               const float* bias, const float* rs,
                               float* oF, int ldF,
                               uint16_t* oB, int ldB, int off) {
  int waves  = (M / 16) * (N / 32);
  int blocks = (waves + 7) / 8;      // 8 waves (256 threads) per block
  gemm_bf16_wmma<<<blocks, 256, 0, st>>>(A, W, M, N, K, mode, scale,
                                         bias, rs, oF, ldF, oB, ldB, off);
}

extern "C" void kernel_launch(void* const* d_in, const int* in_sizes, int n_in,
                              void* d_out, int out_size, void* d_ws, size_t ws_size,
                              hipStream_t stream) {
  (void)in_sizes; (void)n_in; (void)out_size; (void)ws_size;

  const float* x1     = (const float*)d_in[0];
  const float* x2     = (const float*)d_in[1];
  const int*   e1     = (const int*)d_in[2];
  const int*   e2     = (const int*)d_in[3];
  const float* g11Wm  = (const float*)d_in[4];   const float* g11bm = (const float*)d_in[5];
  const float* g11Wn  = (const float*)d_in[6];   const float* g11bn = (const float*)d_in[7];
  const float* g12Wm  = (const float*)d_in[8];   const float* g12bm = (const float*)d_in[9];
  const float* g12Wn  = (const float*)d_in[10];  const float* g12bn = (const float*)d_in[11];
  const float* aff1A  = (const float*)d_in[12];
  const float* cc1W   = (const float*)d_in[13];  const float* cc1b  = (const float*)d_in[14];
  const float* cc2W   = (const float*)d_in[15];  const float* cc2b  = (const float*)d_in[16];
  const float* g21Wm  = (const float*)d_in[17];  const float* g21bm = (const float*)d_in[18];
  const float* g21Wn  = (const float*)d_in[19];  const float* g21bn = (const float*)d_in[20];
  const float* g22Wm  = (const float*)d_in[21];  const float* g22bm = (const float*)d_in[22];
  const float* g22Wn  = (const float*)d_in[23];  const float* g22bn = (const float*)d_in[24];
  const float* aff2A  = (const float*)d_in[25];
  float* out = (float*)d_out;

  // ---- workspace carving ----
  char* base = (char*)d_ws; size_t off = 0;
  auto alloc = [&](size_t bytes) -> void* {
    void* p = base + off; off += (bytes + 255) & ~(size_t)255; return p;
  };
  uint16_t* Mb    = (uint16_t*)alloc((size_t)NN * NN * 2);   // 32 MB affinity matrix
  uint16_t* MTb   = (uint16_t*)alloc((size_t)NN * NN * 2);   // 32 MB transpose
  float* x11  = (float*)alloc((size_t)NN * DH * 4);
  float* x21  = (float*)alloc((size_t)NN * DH * 4);
  float* x13  = (float*)alloc((size_t)NN * DH * 4);
  float* x23  = (float*)alloc((size_t)NN * DH * 4);
  float* Hm   = (float*)alloc((size_t)NN * DH * 4);
  float* Hn   = (float*)alloc((size_t)NN * DH * 4);
  float* agg  = (float*)alloc((size_t)NN * DH * 4);
  float* cnt  = (float*)alloc(NN * 4);
  float* rv   = (float*)alloc(NN * 4);
  float* cv   = (float*)alloc(NN * 4);
  float* tcol = (float*)alloc(NN * 4);
  uint16_t* x1b   = (uint16_t*)alloc((size_t)NN * DIN * 2);
  uint16_t* x2b   = (uint16_t*)alloc((size_t)NN * DIN * 2);
  uint16_t* f1nb  = (uint16_t*)alloc((size_t)NN * DH * 2);
  uint16_t* f2nb  = (uint16_t*)alloc((size_t)NN * DH * 2);
  uint16_t* Tb    = (uint16_t*)alloc((size_t)NN * DH * 2);
  uint16_t* f2cb  = (uint16_t*)alloc((size_t)NN * DH * 2);
  uint16_t* f2cTb = (uint16_t*)alloc((size_t)NN * DH * 2);
  uint16_t* f1rb  = (uint16_t*)alloc((size_t)NN * DH * 2);
  uint16_t* f1rTb = (uint16_t*)alloc((size_t)NN * DH * 2);
  uint16_t* acat1 = (uint16_t*)alloc((size_t)NN * 256 * 2);
  uint16_t* acat2 = (uint16_t*)alloc((size_t)NN * 256 * 2);
  uint16_t* x12b  = (uint16_t*)alloc((size_t)NN * DH * 2);
  uint16_t* x22b  = (uint16_t*)alloc((size_t)NN * DH * 2);
  uint16_t* wm11b = (uint16_t*)alloc(DH * DIN * 2);
  uint16_t* wn11b = (uint16_t*)alloc(DH * DIN * 2);
  uint16_t* wm12b = (uint16_t*)alloc(DH * DIN * 2);
  uint16_t* wn12b = (uint16_t*)alloc(DH * DIN * 2);
  uint16_t* A1b   = (uint16_t*)alloc(DH * DH * 2);
  uint16_t* A2b   = (uint16_t*)alloc(DH * DH * 2);
  uint16_t* cc1Wb = (uint16_t*)alloc(DH * 256 * 2);
  uint16_t* cc2Wb = (uint16_t*)alloc(DH * 256 * 2);
  uint16_t* wm21b = (uint16_t*)alloc(DH * DH * 2);
  uint16_t* wn21b = (uint16_t*)alloc(DH * DH * 2);
  uint16_t* wm22b = (uint16_t*)alloc(DH * DH * 2);
  uint16_t* wn22b = (uint16_t*)alloc(DH * DH * 2);

  auto cvt = [&](const float* src, uint16_t* dst, int n) {
    f32_to_bf16_k<<<(n + 255) / 256, 256, 0, stream>>>(src, dst, n);
  };
  auto fillk = [&](float* p, float v, int n) {
    fill_f32_k<<<(n + 255) / 256, 256, 0, stream>>>(p, v, n);
  };

  // ---- bf16 copies of inputs and weights ----
  cvt(x1, x1b, NN * DIN);      cvt(x2, x2b, NN * DIN);
  cvt(g11Wm, wm11b, DH * DIN); cvt(g11Wn, wn11b, DH * DIN);
  cvt(g12Wm, wm12b, DH * DIN); cvt(g12Wn, wn12b, DH * DIN);
  cvt(aff1A, A1b, DH * DH);    cvt(aff2A, A2b, DH * DH);
  cvt(cc1W, cc1Wb, DH * 256);  cvt(cc2W, cc2Wb, DH * 256);
  cvt(g21Wm, wm21b, DH * DH);  cvt(g21Wn, wn21b, DH * DH);
  cvt(g22Wm, wm22b, DH * DH);  cvt(g22Wn, wn22b, DH * DH);

  // ---- gconv: relu(x Wn^T + bn) + mean_scatter(relu(x Wm^T + bm)) ----
  auto gconv = [&](const uint16_t* xb, int K, const int* edge,
                   const uint16_t* Wmb, const float* bm,
                   const uint16_t* Wnb, const float* bn, float* xout) {
    launch_gemm(stream, xb, Wmb, NN, DH, K, 1, 0.f, bm, nullptr, Hm, DH, nullptr, 0, 0);
    fillk(agg, 0.f, NN * DH);
    fillk(cnt, 0.f, NN);
    scatter_mean_k<<<NE, DH, 0, stream>>>(Hm, edge, agg, cnt, NE, DH);
    launch_gemm(stream, xb, Wnb, NN, DH, K, 1, 0.f, bn, nullptr, Hn, DH, nullptr, 0, 0);
    gconv_combine_k<<<NN, DH, 0, stream>>>(Hn, agg, cnt, xout, DH);
  };

  // ---- affinity -> Mb (bf16), via T = f2n @ A^T then M = exp(clamp(f1n T^T * s)) ----
  auto affinity = [&](const float* f1, const float* f2, const uint16_t* Ab) {
    rownorm_bf16_k<<<NN, DH, 0, stream>>>(f1, f1nb, DH);
    rownorm_bf16_k<<<NN, DH, 0, stream>>>(f2, f2nb, DH);
    launch_gemm(stream, f2nb, Ab, NN, DH, DH, 4, 0.f, nullptr, nullptr,
                nullptr, 0, Tb, DH, 0);
    launch_gemm(stream, f1nb, Tb, NN, NN, DH, 2, 1.0f / (DH * DH), nullptr, nullptr,
                nullptr, 0, Mb, NN, 0);
  };

  // ---- sinkhorn on Mb with scaling vectors rv, cv ----
  auto sinkhorn = [&]() {
    fillk(rv, 1.f, NN);
    fillk(cv, 1.f, NN);
    for (int it = 0; it < SKITERS; ++it) {
      sk_rowpass_k<<<NN, 128, 0, stream>>>(Mb, cv, rv, NN);
      fillk(tcol, 0.f, NN);
      sk_colsum_k<<<dim3(NN / 128, NN / 256), 128, 0, stream>>>(Mb, rv, tcol, NN);
      sk_colfin_k<<<NN / 128, 128, 0, stream>>>(cv, tcol, NN);
    }
  };

  // ================= pipeline =================
  gconv(x1b, DIN, e1, wm11b, g11bm, wn11b, g11bn, x11);
  gconv(x2b, DIN, e2, wm12b, g12bm, wn12b, g12bn, x21);

  affinity(x11, x21, A1b);
  sinkhorn();                                   // rv, cv for M1

  // cross conv: S@f2 = diag(r) M (diag(c) f2) ; S^T@f1 = diag(c) M^T (diag(r) f1)
  transpose_bf16_k<<<dim3(NN / 32, NN / 32), dim3(32, 8), 0, stream>>>(Mb, MTb, NN, NN);

  scale_rows_bf16_k<<<NN, DH, 0, stream>>>(x21, cv, f2cb, DH);
  transpose_bf16_k<<<dim3(DH / 32, NN / 32), dim3(32, 8), 0, stream>>>(f2cb, f2cTb, NN, DH);
  launch_gemm(stream, Mb, f2cTb, NN, DH, NN, 3, 0.f, nullptr, rv,
              nullptr, 0, acat1, 256, 128);            // right half of concat1
  f32_to_bf16_strided_k<<<NN, DH, 0, stream>>>(x11, DH, acat1, 256, 0);

  scale_rows_bf16_k<<<NN, DH, 0, stream>>>(x11, rv, f1rb, DH);
  transpose_bf16_k<<<dim3(DH / 32, NN / 32), dim3(32, 8), 0, stream>>>(f1rb, f1rTb, NN, DH);
  launch_gemm(stream, MTb, f1rTb, NN, DH, NN, 3, 0.f, nullptr, cv,
              nullptr, 0, acat2, 256, 128);            // right half of concat2
  f32_to_bf16_strided_k<<<NN, DH, 0, stream>>>(x21, DH, acat2, 256, 0);

  launch_gemm(stream, acat1, cc1Wb, NN, DH, 256, 6, 0.f, cc1b, nullptr,
              nullptr, 0, x12b, DH, 0);
  launch_gemm(stream, acat2, cc2Wb, NN, DH, 256, 6, 0.f, cc2b, nullptr,
              nullptr, 0, x22b, DH, 0);

  gconv(x12b, DH, e1, wm21b, g21bm, wn21b, g21bn, x13);
  gconv(x22b, DH, e2, wm22b, g22bm, wn22b, g22bn, x23);

  affinity(x13, x23, A2b);
  sinkhorn();                                   // rv, cv for M2

  sk_materialize_k<<<dim3(NN / 128, NN), 128, 0, stream>>>(Mb, rv, cv, out, NN);
}
